// Attention_79121887527548
// MI455X (gfx1250) — matrix operand order repaired and checked by reading, not compiled, not run on hardware
//
#include <hip/hip_runtime.h>
#include <hip/hip_bf16.h>

typedef __attribute__((ext_vector_type(16))) __bf16 v16bf;
typedef __attribute__((ext_vector_type(8)))  float  v8f;
typedef unsigned int v4u __attribute__((ext_vector_type(4)));
typedef int v8i __attribute__((ext_vector_type(8)));
typedef int v4i __attribute__((ext_vector_type(4)));

#define SEQ 4096
#define DIM 128
#define FF  512
#define NB  4                       // batches
#define NROW (NB * SEQ)             // 16384 rows
#define EPS 1e-5f
#define RSCALE 0.08838834764831843f // 1/sqrt(128)
#define KT 64                       // keys per iteration

__device__ __forceinline__ __bf16 f2bf(float f) {
    unsigned u; __builtin_memcpy(&u, &f, 4);
    unsigned r = u + 0x7FFFu + ((u >> 16) & 1u);   // round-to-nearest-even
    unsigned short h = (unsigned short)(r >> 16);
    __bf16 b; __builtin_memcpy(&b, &h, 2);
    return b;
}

__device__ __forceinline__ v8f zero8() {
    v8f z;
#pragma unroll
    for (int i = 0; i < 8; ++i) z[i] = 0.0f;
    return z;
}

// A-fragment K index map for 16-bit 16x32 A matrices (contiguous k = 16*half + e, matching B fills)
__device__ __forceinline__ int amapk(int e, int half) {
    return half * 16 + e;
}

// ---------------------------------------------------------------------------
// DPP16 16-lane butterfly reductions (no LDS, no dscnt waits)
// ---------------------------------------------------------------------------
template <int CTRL>
__device__ __forceinline__ float dpp_mov(float x) {
    int xi = __builtin_bit_cast(int, x);
    int yi = __builtin_amdgcn_update_dpp(xi, xi, CTRL, 0xF, 0xF, true);
    return __builtin_bit_cast(float, yi);
}
__device__ __forceinline__ float red_max16(float x) {
    x = fmaxf(x, dpp_mov<0xB1>(x));    // quad_perm [1,0,3,2]
    x = fmaxf(x, dpp_mov<0x4E>(x));    // quad_perm [2,3,0,1]
    x = fmaxf(x, dpp_mov<0x141>(x));   // row_half_mirror
    x = fmaxf(x, dpp_mov<0x140>(x));   // row_mirror
    return x;
}
__device__ __forceinline__ float red_sum16(float x) {
    x += dpp_mov<0xB1>(x);
    x += dpp_mov<0x4E>(x);
    x += dpp_mov<0x141>(x);
    x += dpp_mov<0x140>(x);
    return x;
}

// ---------------------------------------------------------------------------
// TDM: issue a 2D tensor tile load (global -> LDS).
//   tile: tile_d0 elements (2B) along dim0, tile_d1 rows, row stride d0_stride.
// ---------------------------------------------------------------------------
__device__ __forceinline__ void tdm_load_2d(unsigned lds_addr, const void* gaddr,
                                            unsigned tensor_d0, unsigned tensor_d1,
                                            unsigned tile_d0, unsigned tile_d1,
                                            unsigned d0_stride) {
    unsigned long long ga = (unsigned long long)gaddr;
    v4u g0;
    g0[0] = 1u;                                        // count=1, user desc
    g0[1] = lds_addr;                                  // LDS byte address
    g0[2] = (unsigned)(ga & 0xFFFFFFFFu);              // global addr [31:0]
    g0[3] = (unsigned)((ga >> 32) & 0x01FFFFFFu) | (2u << 30);  // [56:32] | type=2
    v8i g1;
    g1[0] = (int)(1u << 16);                           // data_size=1 (2 bytes)
    g1[1] = (int)((tensor_d0 & 0xFFFFu) << 16);        // tensor_dim0 [15:0]
    g1[2] = (int)(((tensor_d0 >> 16) & 0xFFFFu) | ((tensor_d1 & 0xFFFFu) << 16));
    g1[3] = (int)(((tensor_d1 >> 16) & 0xFFFFu) | ((tile_d0 & 0xFFFFu) << 16));
    g1[4] = (int)(tile_d1 & 0xFFFFu);                  // tile_dim1, tile_dim2=0
    g1[5] = (int)d0_stride;                            // tensor_dim0_stride [31:0]
    g1[6] = 0;                                         // stride [47:32], dim1_stride lo
    g1[7] = 0;
    v4i gz4 = {0, 0, 0, 0};
    v8i gz8 = {0, 0, 0, 0, 0, 0, 0, 0};
    __builtin_amdgcn_tensor_load_to_lds(g0, g1, gz4, gz4, gz8, 0);
}

// ---------------------------------------------------------------------------
// Conversion / transpose kernels (run once; all outputs L2-resident)
// ---------------------------------------------------------------------------
__global__ void cvt_bf16_kernel(const float* __restrict__ in,
                                __bf16* __restrict__ out, int n, float scale) {
    int i = blockIdx.x * blockDim.x + threadIdx.x;
    if (i < n) out[i] = f2bf(in[i] * scale);
}

// V [b][s][d] f32 -> Vt [b][d][s] bf16 (coalesced writes)
__global__ void cvt_vt_kernel(const float* __restrict__ v,
                              __bf16* __restrict__ vt) {
    int i = blockIdx.x * blockDim.x + threadIdx.x;   // i = ((b*DIM)+d)*SEQ + s
    int b = i / (DIM * SEQ);
    int rem = i - b * DIM * SEQ;
    int d = rem / SEQ;
    int s = rem - d * SEQ;
    vt[i] = f2bf(v[(size_t)(b * SEQ + s) * DIM + d]);
}

// w1 [D][F] f32 -> w1t [F][D] bf16
__global__ void cvt_w1t_kernel(const float* __restrict__ w1,
                               __bf16* __restrict__ w1t) {
    int i = blockIdx.x * blockDim.x + threadIdx.x;   // i = f*DIM + d
    int f = i / DIM, d = i - f * DIM;
    w1t[i] = f2bf(w1[(size_t)d * FF + f]);
}

// w2 [F][D] f32 -> w2t [D][F] bf16
__global__ void cvt_w2t_kernel(const float* __restrict__ w2,
                               __bf16* __restrict__ w2t) {
    int i = blockIdx.x * blockDim.x + threadIdx.x;   // i = d*FF + f
    int d = i / FF, f = i - d * FF;
    w2t[i] = f2bf(w2[(size_t)f * DIM + d]);
}

// ---------------------------------------------------------------------------
// Flash attention. 4 waves/block, each owns a 16-query tile; all share the
// V tiles, which are DMA'd (TDM) into double-buffered LDS by wave 0 and
// overlapped with the QK^T WMMA + softmax phase. K streams straight to VGPRs.
// ---------------------------------------------------------------------------
__global__ __launch_bounds__(128) void flash_attn_kernel(
    const __bf16* __restrict__ qb, const __bf16* __restrict__ kb,
    const __bf16* __restrict__ vt, float* __restrict__ attn_out) {

    __shared__ __attribute__((aligned(16))) __bf16 vtile[2][DIM * KT];  // 2 x 16KB
    __shared__ __attribute__((aligned(16))) __bf16 lds_p[4][16 * KT];   // P staging

    const int wave = threadIdx.x >> 5;
    const int lane = threadIdx.x & 31;
    const int half = lane >> 4;
    const int ln   = lane & 15;

    const int tile = blockIdx.x * 4 + wave;       // 0..1023 (16-query tiles)
    const int b    = tile >> 8;                   // 256 tiles per batch
    const int q0   = (tile & 255) << 4;

    const __bf16* Q  = qb + (size_t)b * SEQ * DIM;
    const __bf16* K  = kb + (size_t)b * SEQ * DIM;
    const __bf16* VT = vt + (size_t)b * DIM * SEQ;   // [d][s]
    __bf16* pbuf = lds_p[wave];

    // Q A-fragments: 4 slices of K=32 over D=128 (lane holds row m = ln)
    v16bf qa[4];
#pragma unroll
    for (int s = 0; s < 4; ++s)
#pragma unroll
        for (int e = 0; e < 16; ++e)
            qa[s][e] = Q[(size_t)(q0 + ln) * DIM + s * 32 + amapk(e, half)];

    v8f o[8];
#pragma unroll
    for (int i = 0; i < 8; ++i) o[i] = zero8();
    float rmax[8], rsum[8];
#pragma unroll
    for (int r = 0; r < 8; ++r) { rmax[r] = -3.0e38f; rsum[r] = 0.0f; }

    // Prologue: DMA first V tile (VT[0..127][0..63]) into buffer 0
    if (wave == 0)
        tdm_load_2d((unsigned)(unsigned long long)&vtile[0][0], VT,
                    SEQ, DIM, KT, DIM, SEQ);

    for (int kt = 0; kt < SEQ; kt += KT) {
        const int sel = (kt >> 6) & 1;
        // Prefetch next V tile into the other buffer (overlaps QK + softmax)
        if (wave == 0 && kt + KT < SEQ)
            tdm_load_2d((unsigned)(unsigned long long)&vtile[sel ^ 1][0], VT + kt + KT,
                        SEQ, DIM, KT, DIM, SEQ);

        // ---- logits: Q(16x128) x K^T -> four 16x16 tiles (keys kt+g*16..+15)
        v8f c[4];
#pragma unroll
        for (int g = 0; g < 4; ++g) c[g] = zero8();
#pragma unroll
        for (int s = 0; s < 4; ++s) {
#pragma unroll
            for (int g = 0; g < 4; ++g) {
                v16bf bk;
                const __bf16* Kp = K + (size_t)(kt + g * 16 + ln) * DIM + s * 32 + half * 16;
#pragma unroll
                for (int e = 0; e < 16; ++e) bk[e] = Kp[e];
                c[g] = __builtin_amdgcn_wmma_f32_16x16x32_bf16(false, qa[s], false, bk,
                                                               (short)0, c[g], false, false);
            }
        }

        // ---- online softmax (rows m = r + 8*half live in 16-lane half-groups)
#pragma unroll
        for (int r = 0; r < 8; ++r) {
            float l0 = c[0][r], l1 = c[1][r], l2 = c[2][r], l3 = c[3][r];
            float tm = fmaxf(fmaxf(l0, l1), fmaxf(l2, l3));
            float nm   = fmaxf(rmax[r], red_max16(tm));
            float corr = __expf(rmax[r] - nm);
            float p0 = __expf(l0 - nm), p1 = __expf(l1 - nm);
            float p2 = __expf(l2 - nm), p3 = __expf(l3 - nm);
            rsum[r] = rsum[r] * corr + red_sum16((p0 + p1) + (p2 + p3));
            rmax[r] = nm;
#pragma unroll
            for (int nb = 0; nb < 8; ++nb) o[nb][r] *= corr;
            __bf16* pr = pbuf + (r + 8 * half) * KT + ln;
            pr[0]  = f2bf(p0);
            pr[16] = f2bf(p1);
            pr[32] = f2bf(p2);
            pr[48] = f2bf(p3);
        }
        asm volatile("s_wait_dscnt 0" ::: "memory");   // intra-wave LDS transpose

        // ---- P A-fragments (16x64, contraction over the 64 keys)
        v16bf pa0, pa1;
#pragma unroll
        for (int e = 0; e < 16; ++e) {
            pa0[e] = pbuf[ln * KT + amapk(e, half)];
            pa1[e] = pbuf[ln * KT + 32 + amapk(e, half)];
        }

        // ---- current V tile must be resident; sync the workgroup
        if (wave == 0) {
            if (kt + KT < SEQ) __builtin_amdgcn_s_wait_tensorcnt(1);  // prefetch may fly
            else               __builtin_amdgcn_s_wait_tensorcnt(0);
        }
        __syncthreads();

        // ---- O += P x V  (2x8 WMMAs; V B-frags from the shared LDS tile)
        const __bf16* vb = &vtile[sel][0];
#pragma unroll
        for (int nb = 0; nb < 8; ++nb) {
            v16bf vf0, vf1;
            const __bf16* Vp = vb + (nb * 16 + ln) * KT + half * 16;
#pragma unroll
            for (int e = 0; e < 16; ++e) { vf0[e] = Vp[e]; vf1[e] = Vp[32 + e]; }
            o[nb] = __builtin_amdgcn_wmma_f32_16x16x32_bf16(false, pa0, false, vf0,
                                                            (short)0, o[nb], false, false);
            o[nb] = __builtin_amdgcn_wmma_f32_16x16x32_bf16(false, pa1, false, vf1,
                                                            (short)0, o[nb], false, false);
        }
        __syncthreads();   // all waves done reading before this buffer is reloaded
    }

    // ---- epilogue: divide by row sums, write f32
#pragma unroll
    for (int r = 0; r < 8; ++r) {
        float inv = 1.0f / rsum[r];
#pragma unroll
        for (int nb = 0; nb < 8; ++nb) o[nb][r] *= inv;
    }
    float* O = attn_out + (size_t)b * SEQ * DIM;
#pragma unroll
    for (int nb = 0; nb < 8; ++nb)
#pragma unroll
        for (int r = 0; r < 8; ++r)
            O[(size_t)(q0 + r + 8 * half) * DIM + nb * 16 + ln] = o[nb][r];
}

// ---------------------------------------------------------------------------
// Fused LN1 + FFN (WMMA bf16) + residual + LN2. One wave per 16-row tile.
// ---------------------------------------------------------------------------
__global__ __launch_bounds__(64) void ffn_kernel(
    const float* __restrict__ attn_out,
    const __bf16* __restrict__ w1t, const __bf16* __restrict__ w2t,
    const float* __restrict__ b1, const float* __restrict__ b2,
    const float* __restrict__ g1, const float* __restrict__ be1,
    const float* __restrict__ g2, const float* __restrict__ be2,
    float* __restrict__ out) {

    __shared__ __attribute__((aligned(16))) __bf16 xbuf[2][16 * DIM];
    __shared__ __attribute__((aligned(16))) __bf16 hbuf[2][16 * FF];

    const int wave = threadIdx.x >> 5;
    const int lane = threadIdx.x & 31;
    const int half = lane >> 4;
    const int ln   = lane & 15;
    const int row0 = (blockIdx.x * 2 + wave) * 16;

    // ---- load x tile in C-layout regs: x[dblk][r] = x[m=r+8*half][d=dblk*16+ln]
    float x[8][8];
#pragma unroll
    for (int dblk = 0; dblk < 8; ++dblk)
#pragma unroll
        for (int r = 0; r < 8; ++r)
            x[dblk][r] = attn_out[(size_t)(row0 + r + 8 * half) * DIM + dblk * 16 + ln];

    // ---- LN1 (per-row reductions over 8 regs x 16 lanes); overwrite x with x-hat
#pragma unroll
    for (int r = 0; r < 8; ++r) {
        float s = 0.0f, s2 = 0.0f;
#pragma unroll
        for (int dblk = 0; dblk < 8; ++dblk) { s += x[dblk][r]; s2 += x[dblk][r] * x[dblk][r]; }
        s = red_sum16(s); s2 = red_sum16(s2);
        float mean = s * (1.0f / DIM);
        float var  = s2 * (1.0f / DIM) - mean * mean;
        float rstd = rsqrtf(var + EPS);
#pragma unroll
        for (int dblk = 0; dblk < 8; ++dblk) {
            int d = dblk * 16 + ln;
            float v = (x[dblk][r] - mean) * rstd * g1[d] + be1[d];
            x[dblk][r] = v;
            xbuf[wave][(r + 8 * half) * DIM + d] = f2bf(v);
        }
    }
    asm volatile("s_wait_dscnt 0" ::: "memory");

    // ---- x-hat A-fragments (16x128 -> 4 slices)
    v16bf xa[4];
#pragma unroll
    for (int s = 0; s < 4; ++s)
#pragma unroll
        for (int e = 0; e < 16; ++e)
            xa[s][e] = xbuf[wave][ln * DIM + s * 32 + amapk(e, half)];

    // ---- h = relu(x @ w1 + b1): 32 N-blocks of 16 over F=512
    for (int nblk = 0; nblk < 32; ++nblk) {
        v8f acc = zero8();
#pragma unroll
        for (int s = 0; s < 4; ++s) {
            v16bf bw;
            const __bf16* Wp = w1t + (size_t)(nblk * 16 + ln) * DIM + s * 32 + half * 16;
#pragma unroll
            for (int e = 0; e < 16; ++e) bw[e] = Wp[e];
            acc = __builtin_amdgcn_wmma_f32_16x16x32_bf16(false, xa[s], false, bw,
                                                          (short)0, acc, false, false);
        }
        float bias = b1[nblk * 16 + ln];
#pragma unroll
        for (int r = 0; r < 8; ++r) {
            float h = fmaxf(acc[r] + bias, 0.0f);
            hbuf[wave][(r + 8 * half) * FF + nblk * 16 + ln] = f2bf(h);
        }
    }
    asm volatile("s_wait_dscnt 0" ::: "memory");

    // ---- y = h @ w2: contraction F=512 in 16 slices of 32, N = 8 blocks of 16
    v8f y[8];
#pragma unroll
    for (int i = 0; i < 8; ++i) y[i] = zero8();
    for (int s = 0; s < 16; ++s) {
        v16bf ha;
#pragma unroll
        for (int e = 0; e < 16; ++e)
            ha[e] = hbuf[wave][ln * FF + s * 32 + amapk(e, half)];
#pragma unroll
        for (int dblk = 0; dblk < 8; ++dblk) {
            v16bf bw;
            const __bf16* Wp = w2t + (size_t)(dblk * 16 + ln) * FF + s * 32 + half * 16;
#pragma unroll
            for (int e = 0; e < 16; ++e) bw[e] = Wp[e];
            y[dblk] = __builtin_amdgcn_wmma_f32_16x16x32_bf16(false, ha, false, bw,
                                                              (short)0, y[dblk], false, false);
        }
    }

    // ---- residual + LN2, write output
#pragma unroll
    for (int r = 0; r < 8; ++r) {
        float res[8];
        float s = 0.0f, s2 = 0.0f;
#pragma unroll
        for (int dblk = 0; dblk < 8; ++dblk) {
            float v = x[dblk][r] + y[dblk][r] + b2[dblk * 16 + ln];
            res[dblk] = v; s += v; s2 += v * v;
        }
        s = red_sum16(s); s2 = red_sum16(s2);
        float mean = s * (1.0f / DIM);
        float var  = s2 * (1.0f / DIM) - mean * mean;
        float rstd = rsqrtf(var + EPS);
#pragma unroll
        for (int dblk = 0; dblk < 8; ++dblk) {
            int d = dblk * 16 + ln;
            out[(size_t)(row0 + r + 8 * half) * DIM + d] = (res[dblk] - mean) * rstd * g2[d] + be2[d];
        }
    }
}

// ---------------------------------------------------------------------------
extern "C" void kernel_launch(void* const* d_in, const int* in_sizes, int n_in,
                              void* d_out, int out_size, void* d_ws, size_t ws_size,
                              hipStream_t stream) {
    const float* q  = (const float*)d_in[0];
    const float* k  = (const float*)d_in[1];
    const float* v  = (const float*)d_in[2];
    const float* w1 = (const float*)d_in[3];
    const float* b1 = (const float*)d_in[4];
    const float* w2 = (const float*)d_in[5];
    const float* b2 = (const float*)d_in[6];
    const float* g1 = (const float*)d_in[7];
    const float* be1 = (const float*)d_in[8];
    const float* g2 = (const float*)d_in[9];
    const float* be2 = (const float*)d_in[10];
    float* out = (float*)d_out;

    const int NQKV = NB * SEQ * DIM;   // 2,097,152
    const int NW1  = DIM * FF;         // 65,536
    const int NW2  = FF * DIM;         // 65,536

    char* ws = (char*)d_ws;
    __bf16* qb  = (__bf16*)(ws);
    __bf16* kb  = (__bf16*)(ws + (size_t)NQKV * 2);
    __bf16* vtb = (__bf16*)(ws + (size_t)NQKV * 4);
    __bf16* w1t = (__bf16*)(ws + (size_t)NQKV * 6);
    __bf16* w2t = (__bf16*)(ws + (size_t)NQKV * 6 + (size_t)NW1 * 2);
    float* attn = (float*)(ws + (size_t)NQKV * 6 + (size_t)NW1 * 2 + (size_t)NW2 * 2);

    // Q gets 1/sqrt(D) folded into the bf16 conversion
    cvt_bf16_kernel<<<(NQKV + 255) / 256, 256, 0, stream>>>(q, qb, NQKV, RSCALE);
    cvt_bf16_kernel<<<(NQKV + 255) / 256, 256, 0, stream>>>(k, kb, NQKV, 1.0f);
    cvt_vt_kernel<<<NQKV / 256, 256, 0, stream>>>(v, vtb);
    cvt_w1t_kernel<<<NW1 / 256, 256, 0, stream>>>(w1, w1t);
    cvt_w2t_kernel<<<NW2 / 256, 256, 0, stream>>>(w2, w2t);

    // 1024 query tiles of 16 rows, 4 waves (128 threads) per block
    flash_attn_kernel<<<NB * (SEQ / 16) / 4, 128, 0, stream>>>(qb, kb, vtb, attn);

    // 1024 row tiles of 16, 2 waves (64 threads) per block
    ffn_kernel<<<NROW / 16 / 2, 64, 0, stream>>>(attn, w1t, w2t, b1, b2,
                                                 g1, be1, g2, be2, out);
}
